// MultiHeadSA_12756052869505
// MI455X (gfx1250) — matrix operand order repaired and checked by reading, not compiled, not run on hardware
//
#include <hip/hip_runtime.h>
#include <hip/hip_bf16.h>

typedef __attribute__((ext_vector_type(16))) _Float16 v16h;
typedef __attribute__((ext_vector_type(8)))  float    v8f;
typedef unsigned int u32;
typedef __attribute__((ext_vector_type(4))) u32 u32x4;
typedef __attribute__((ext_vector_type(8))) int i32x8;
typedef __attribute__((ext_vector_type(4))) int i32x4;

#define D_MODEL 1024
#define N_HEADS 16
#define DHEAD   64
#define SEQ     2048
#define BATCH   4
#define QKV_STRIDE ((size_t)BATCH * N_HEADS * SEQ * DHEAD)   // 8M elements

// Tensor Data Mover path (this toolchain: 6-arg builtin, clang-23/therock)
#ifndef USE_TDM
#if defined(__has_builtin)
#if __has_builtin(__builtin_amdgcn_tensor_load_to_lds) && __has_builtin(__builtin_amdgcn_s_wait_tensorcnt)
#define USE_TDM 1
#else
#define USE_TDM 0
#endif
#else
#define USE_TDM 0
#endif
#endif

union FragA { v16h h; uint4 u[2]; };

static __device__ inline v8f wmma_f16(v16h a, v16h b, v8f c) {
  // D = A(16x32 f16) * B(32x16 f16) + C(16x16 f32)
  return __builtin_amdgcn_wmma_f32_16x16x32_f16(false, a, false, b, (short)0, c,
                                                false, false);
}

static __device__ inline v8f vzero8() { v8f z = {0.f,0.f,0.f,0.f,0.f,0.f,0.f,0.f}; return z; }

// ---------------------------------------------------------------------------
// Kernel 0: RoPE cos/sin LUT:  lut[s*32+p] = {cos(pos[s]*f_p), sin(pos[s]*f_p)}
// ---------------------------------------------------------------------------
__global__ void __launch_bounds__(256)
rope_lut_kernel(const int* __restrict__ pos, float2* __restrict__ lut)
{
  const float NEG_LOG_TH_64 = 0.14391156831f;      // ln(10000)/64
  int idx = blockIdx.x * 256 + threadIdx.x;        // 2048*32 entries
  int si = idx >> 5, p = idx & 31;
  float ang = (float)pos[si] * __expf(-(float)(2 * p) * NEG_LOG_TH_64);
  float sn, cs; __sincosf(ang, &sn, &cs);
  lut[idx] = make_float2(cs, sn);
}

// ---------------------------------------------------------------------------
// Kernel 1: qkv = x @ W_qkv^T, fused RoPE on q,k, scatter to [3][B,H,S,64] f16
// M = B*S = 8192, N = 3*D = 3072, K = D = 1024
// ---------------------------------------------------------------------------
__global__ void __launch_bounds__(256)
qkv_rope_kernel(const float* __restrict__ X, const float* __restrict__ W,
                const float2* __restrict__ lut, _Float16* __restrict__ qkvh)
{
  __shared__ __align__(16) _Float16 As[128][40];   // [m][k]
  __shared__ __align__(16) _Float16 Bs[128][40];   // [n][k]
  const int tid  = threadIdx.x;
  const int wave = tid >> 5, lane = tid & 31;
  const int m0 = blockIdx.y * 128;
  const int n0 = blockIdx.x * 128;
  const int waveM = (wave & 3) * 32;
  const int waveN = (wave >> 2) * 64;
  const int arow = lane & 15;
  const int kgrp = (lane >> 4) * 8;    // A-frag K base (runs at kgrp, kgrp+16)
  const int kb16 = (lane >> 4) * 16;   // unused; B frags also load runs at kgrp, kgrp+16

  v8f acc[2][4];
  for (int mt = 0; mt < 2; ++mt)
    for (int nt = 0; nt < 4; ++nt) acc[mt][nt] = vzero8();

  for (int k0 = 0; k0 < D_MODEL; k0 += 32) {
    for (int it = 0; it < 4; ++it) {               // A tile: 128x32 f32 -> f16
      int li = tid + it * 256;
      int r = li >> 3, c4 = (li & 7) * 4;
      float4 xv = *(const float4*)&X[(size_t)(m0 + r) * D_MODEL + k0 + c4];
      As[r][c4+0] = (_Float16)xv.x; As[r][c4+1] = (_Float16)xv.y;
      As[r][c4+2] = (_Float16)xv.z; As[r][c4+3] = (_Float16)xv.w;
    }
    for (int it = 0; it < 4; ++it) {               // B tile: 128x32 f32 -> f16
      int li = tid + it * 256;
      int r = li >> 3, c4 = (li & 7) * 4;
      float4 wv = *(const float4*)&W[(size_t)(n0 + r) * D_MODEL + k0 + c4];
      Bs[r][c4+0] = (_Float16)wv.x; Bs[r][c4+1] = (_Float16)wv.y;
      Bs[r][c4+2] = (_Float16)wv.z; Bs[r][c4+3] = (_Float16)wv.w;
    }
    __syncthreads();
    FragA a[2], b[4];
    for (int mt = 0; mt < 2; ++mt) {
      const _Float16* p = &As[waveM + mt*16 + arow][kgrp];
      a[mt].u[0] = *(const uint4*)p;
      a[mt].u[1] = *(const uint4*)(p + 16);
    }
    for (int nt = 0; nt < 4; ++nt) {
      const _Float16* p = &Bs[waveN + nt*16 + (lane & 15)][kgrp];
      b[nt].u[0] = *(const uint4*)p;
      b[nt].u[1] = *(const uint4*)(p + 16);
    }
    for (int mt = 0; mt < 2; ++mt)
      for (int nt = 0; nt < 4; ++nt)
        acc[mt][nt] = wmma_f16(a[mt].h, b[nt].h, acc[mt][nt]);
    __syncthreads();
  }

  // epilogue: branchless RoPE via LUT + scatter into contiguous [3][B,H,S,64]
  for (int mt = 0; mt < 2; ++mt) {
    int grow = m0 + waveM + mt*16 + ((lane < 16) ? 0 : 8);
    for (int nt = 0; nt < 4; ++nt) {
      int gcol = n0 + waveN + nt*16 + (lane & 15);
      int which = gcol >> 10;                      // 0=q 1=k 2=v
      int d = gcol & (D_MODEL - 1);
      int head = d >> 6, dh = d & 63;
      for (int r = 0; r < 8; ++r) {
        float val = acc[mt][nt][r];
        int row = grow + r;
        int bi = row >> 11, si = row & (SEQ - 1);
        float other = __shfl_xor(val, 1, 32);      // pair partner (dh ^ 1)
        float2 cs = lut[si * 32 + (dh >> 1)];      // always valid address
        float roped = ((dh & 1) == 0) ? (val * cs.x - other * cs.y)
                                      : (other * cs.y + val * cs.x);
        float outv = (which < 2) ? roped : val;    // v_cndmask, no branch
        size_t di = (size_t)which * QKV_STRIDE +
                    ((size_t)(bi*N_HEADS + head)*SEQ + si)*DHEAD + dh;
        qkvh[di] = (_Float16)outv;
      }
    }
  }
}

// ---------------------------------------------------------------------------
// Kernel 2: causal flash attention per (b, h, 64-query tile); 4 waves/block
// ---------------------------------------------------------------------------
__global__ void __launch_bounds__(128)
attn_kernel(const _Float16* __restrict__ qh, const _Float16* __restrict__ kh,
            const _Float16* __restrict__ vh, _Float16* __restrict__ oh)
{
  __shared__ __align__(16) _Float16 Ks[64][72];      // [key][dh]  (= B layout of K^T)
  __shared__ __align__(16) _Float16 Vs[64][72];      // [dh][key]  (= B layout of V)
  __shared__ __align__(16) _Float16 Ps[4][16][72];   // per-wave P tile
  const int tid  = threadIdx.x;
  const int wave = tid >> 5, lane = tid & 31;
  const int arow = lane & 15;
  const int kgrp = (lane >> 4) * 8;
  const int kb16 = (lane >> 4) * 16;
  const int bh = blockIdx.y;
  const int b = bh >> 4, h = bh & 15;
  const int q0 = blockIdx.x * 64;
  const size_t base = ((size_t)b * N_HEADS + h) * SEQ;

  FragA aq[2];                                       // persistent Q frags (16x64)
  {
    const _Float16* qp = &qh[(base + q0 + wave*16 + arow) * DHEAD];
    for (int kk = 0; kk < 2; ++kk) {
      aq[kk].u[0] = *(const uint4*)(qp + kk*32 + kgrp);
      aq[kk].u[1] = *(const uint4*)(qp + kk*32 + kgrp + 16);
    }
  }

  float m_i[8], l_i[8];
  for (int r = 0; r < 8; ++r) { m_i[r] = -1e30f; l_i[r] = 0.f; }
  v8f o[4];
  for (int nt = 0; nt < 4; ++nt) o[nt] = vzero8();

  for (int kb = 0; kb <= (int)blockIdx.x; ++kb) {
#if USE_TDM
    // --- K tile via Tensor Data Mover: global [64][64] f16 -> LDS stride-72 ---
    if (wave == 0) {
      const _Float16* ksrc = &kh[(base + kb*64) * DHEAD];
      unsigned long long ga = (unsigned long long)(uintptr_t)ksrc;
      u32 lds_base = (u32)(uintptr_t)&Ks[0][0];
      u32x4 g0;
      g0[0] = 1u;                                     // count=1 (valid D#)
      g0[1] = lds_base;                               // lds_addr
      g0[2] = (u32)ga;                                // global_addr[31:0]
      g0[3] = ((u32)(ga >> 32) & 0x01FFFFFFu) | (2u << 30);  // ga[56:32] | type=2
      i32x8 g1;
      g1[0] = (1 << 16)      // data_size = 2 bytes
            | (1 << 20)      // pad_enable
            | (4 << 22)      // pad_interval: every 32 DWORDs (=64 halves)
            | (3 << 25);     // pad_amount: 4 DWORDs (=8 halves) -> stride 72
      g1[1] = (64 << 16);    // tensor_dim0[15:0]=64
      g1[2] = (64 << 16);    // tensor_dim1[15:0]=64
      g1[3] = (64 << 16);    // tile_dim0=64
      g1[4] = 64;            // tile_dim1=64
      g1[5] = 64;            // tensor_dim0_stride=64
      g1[6] = 0;
      g1[7] = 0;
      i32x4 gz4 = {0, 0, 0, 0};
      i32x8 gz8 = {0, 0, 0, 0, 0, 0, 0, 0};
      __builtin_amdgcn_tensor_load_to_lds(g0, g1, gz4, gz4, gz8, 0);
    }
    for (int it = 0; it < 4; ++it) {                 // V transposed (manual)
      int li = tid + it * 128;
      int key = li >> 3, c8 = (li & 7) * 8;
      union { uint4 u; _Float16 e[8]; } vv;
      vv.u = *(const uint4*)&vh[(base + kb*64 + key)*DHEAD + c8];
      for (int j = 0; j < 8; ++j) Vs[c8 + j][key] = vv.e[j];
    }
    if (wave == 0) __builtin_amdgcn_s_wait_tensorcnt(0);
#else
    for (int it = 0; it < 4; ++it) {                 // stage K raw, V transposed
      int li = tid + it * 128;
      int key = li >> 3, c8 = (li & 7) * 8;
      *(uint4*)&Ks[key][c8] =
          *(const uint4*)&kh[(base + kb*64 + key)*DHEAD + c8];
      union { uint4 u; _Float16 e[8]; } vv;
      vv.u = *(const uint4*)&vh[(base + kb*64 + key)*DHEAD + c8];
      for (int j = 0; j < 8; ++j) Vs[c8 + j][key] = vv.e[j];
    }
#endif
    __syncthreads();

    // S = Q K^T  (16x64 per wave)
    v8f s[4];
    for (int nt = 0; nt < 4; ++nt) s[nt] = vzero8();
    for (int kk = 0; kk < 2; ++kk)
      for (int nt = 0; nt < 4; ++nt) {
        FragA bk;
        const _Float16* p = &Ks[nt*16 + (lane & 15)][kk*32 + kgrp];
        bk.u[0] = *(const uint4*)p;
        bk.u[1] = *(const uint4*)(p + 16);
        s[nt] = wmma_f16(aq[kk].h, bk.h, s[nt]);
      }

    // causal mask + scale + online softmax
    float sm[4][8];
    const int qrow_base = q0 + wave*16 + ((lane < 16) ? 0 : 8);
    for (int nt = 0; nt < 4; ++nt) {
      int keyg = kb*64 + nt*16 + (lane & 15);
      for (int r = 0; r < 8; ++r) {
        float v = s[nt][r] * 0.125f;                 // 1/sqrt(64)
        if (keyg > qrow_base + r) v = -1e30f;
        sm[nt][r] = v;
      }
    }
    float sc[8];
    for (int r = 0; r < 8; ++r) {
      float rm = fmaxf(fmaxf(sm[0][r], sm[1][r]), fmaxf(sm[2][r], sm[3][r]));
      for (int off = 1; off <= 8; off <<= 1)
        rm = fmaxf(rm, __shfl_xor(rm, off, 32));     // reduce across 16-lane half
      float mn = fmaxf(m_i[r], rm);
      sc[r] = __expf(m_i[r] - mn);
      float rs = 0.f;
      for (int nt = 0; nt < 4; ++nt) {
        float p = __expf(sm[nt][r] - mn);
        sm[nt][r] = p;
        rs += p;
      }
      for (int off = 1; off <= 8; off <<= 1)
        rs += __shfl_xor(rs, off, 32);
      l_i[r] = l_i[r] * sc[r] + rs;
      m_i[r] = mn;
    }
    for (int nt = 0; nt < 4; ++nt) {                 // P -> LDS; rescale O
      int col = nt*16 + (lane & 15);
      for (int r = 0; r < 8; ++r) {
        Ps[wave][(lane < 16) ? r : r + 8][col] = (_Float16)sm[nt][r];
        o[nt][r] *= sc[r];
      }
    }
    // LDS ops are in-order within a wave; just drain DScnt (no barrier needed)
    asm volatile("s_wait_dscnt 0" ::: "memory");

    // O += P V
    for (int kk = 0; kk < 2; ++kk) {
      FragA ap;
      const _Float16* pp = &Ps[wave][arow][kk*32 + kgrp];
      ap.u[0] = *(const uint4*)pp;
      ap.u[1] = *(const uint4*)(pp + 16);
      for (int nt = 0; nt < 4; ++nt) {
        FragA bv;
        const _Float16* vp = &Vs[nt*16 + (lane & 15)][kk*32 + kgrp];
        bv.u[0] = *(const uint4*)vp;
        bv.u[1] = *(const uint4*)(vp + 16);
        o[nt] = wmma_f16(ap.h, bv.h, o[nt]);
      }
    }
    __syncthreads();
  }

  for (int nt = 0; nt < 4; ++nt) {                   // O / l -> [B,S,D] f16
    int col = h*DHEAD + nt*16 + (lane & 15);
    for (int r = 0; r < 8; ++r) {
      int rowg = q0 + wave*16 + ((lane < 16) ? r : r + 8);
      float outv = o[nt][r] / l_i[r];
      oh[((size_t)b * SEQ + rowg) * D_MODEL + col] = (_Float16)outv;
    }
  }
}

// ---------------------------------------------------------------------------
// Kernel 3: out = O @ W_out^T   (M=8192, N=1024, K=1024), f32 output
// ---------------------------------------------------------------------------
__global__ void __launch_bounds__(256)
out_proj_kernel(const _Float16* __restrict__ A, const float* __restrict__ W,
                float* __restrict__ out)
{
  __shared__ __align__(16) _Float16 As[128][40];
  __shared__ __align__(16) _Float16 Bs[128][40];
  const int tid  = threadIdx.x;
  const int wave = tid >> 5, lane = tid & 31;
  const int m0 = blockIdx.y * 128;
  const int n0 = blockIdx.x * 128;
  const int waveM = (wave & 3) * 32;
  const int waveN = (wave >> 2) * 64;
  const int arow = lane & 15;
  const int kgrp = (lane >> 4) * 8;
  const int kb16 = (lane >> 4) * 16;

  v8f acc[2][4];
  for (int mt = 0; mt < 2; ++mt)
    for (int nt = 0; nt < 4; ++nt) acc[mt][nt] = vzero8();

  for (int k0 = 0; k0 < D_MODEL; k0 += 32) {
    for (int it = 0; it < 2; ++it) {                 // A tile: f16 direct
      int li = tid + it * 256;
      int r = li >> 2, c8 = (li & 3) * 8;
      *(uint4*)&As[r][c8] =
          *(const uint4*)&A[(size_t)(m0 + r) * D_MODEL + k0 + c8];
    }
    for (int it = 0; it < 4; ++it) {                 // B tile: f32 -> f16
      int li = tid + it * 256;
      int r = li >> 3, c4 = (li & 7) * 4;
      float4 wv = *(const float4*)&W[(size_t)(n0 + r) * D_MODEL + k0 + c4];
      Bs[r][c4+0] = (_Float16)wv.x; Bs[r][c4+1] = (_Float16)wv.y;
      Bs[r][c4+2] = (_Float16)wv.z; Bs[r][c4+3] = (_Float16)wv.w;
    }
    __syncthreads();
    FragA a[2], b[4];
    for (int mt = 0; mt < 2; ++mt) {
      const _Float16* p = &As[waveM + mt*16 + arow][kgrp];
      a[mt].u[0] = *(const uint4*)p;
      a[mt].u[1] = *(const uint4*)(p + 16);
    }
    for (int nt = 0; nt < 4; ++nt) {
      const _Float16* p = &Bs[waveN + nt*16 + (lane & 15)][kgrp];
      b[nt].u[0] = *(const uint4*)p;
      b[nt].u[1] = *(const uint4*)(p + 16);
    }
    for (int mt = 0; mt < 2; ++mt)
      for (int nt = 0; nt < 4; ++nt)
        acc[mt][nt] = wmma_f16(a[mt].h, b[nt].h, acc[mt][nt]);
    __syncthreads();
  }

  for (int mt = 0; mt < 2; ++mt) {
    int grow = m0 + waveM + mt*16 + ((lane < 16) ? 0 : 8);
    for (int nt = 0; nt < 4; ++nt) {
      int gcol = n0 + waveN + nt*16 + (lane & 15);
      for (int r = 0; r < 8; ++r)
        out[(size_t)(grow + r) * D_MODEL + gcol] = acc[mt][nt][r];
    }
  }
}

// ---------------------------------------------------------------------------
extern "C" void kernel_launch(void* const* d_in, const int* in_sizes, int n_in,
                              void* d_out, int out_size, void* d_ws, size_t ws_size,
                              hipStream_t stream) {
  const float* x    = (const float*)d_in[0];
  const int*   pos  = (const int*)d_in[1];
  const float* wqkv = (const float*)d_in[2];
  const float* wout = (const float*)d_in[3];
  float* out = (float*)d_out;

  _Float16* ws = (_Float16*)d_ws;
  const size_t per = QKV_STRIDE;                              // 8M f16 = 16 MB
  _Float16* qh = ws;
  _Float16* kh = ws + per;
  _Float16* vh = ws + 2 * per;
  _Float16* oh = ws + 3 * per;
  float2* lut  = (float2*)(ws + 4 * per);                     // 2048*32 float2

  rope_lut_kernel<<<dim3((SEQ * 32) / 256), dim3(256), 0, stream>>>(pos, lut);

  dim3 g1(3 * D_MODEL / 128, (BATCH * SEQ) / 128);
  qkv_rope_kernel<<<g1, dim3(256), 0, stream>>>(x, wqkv, lut, qh);

  dim3 g2(SEQ / 64, BATCH * N_HEADS);
  attn_kernel<<<g2, dim3(128), 0, stream>>>(qh, kh, vh, oh);

  dim3 g3(D_MODEL / 128, (BATCH * SEQ) / 128);
  out_proj_kernel<<<g3, dim3(256), 0, stream>>>(oh, wout, out);
}
